// VGAEEncoder_10685878632449
// MI455X (gfx1250) — compile-verified, matrix operand order checked by reading, not run on hardware
//
#include <hip/hip_runtime.h>
#include <hip/hip_bf16.h>

typedef __attribute__((ext_vector_type(2)))  float    v2f;
typedef __attribute__((ext_vector_type(8)))  float    v8f;
typedef __attribute__((ext_vector_type(16))) _Float16 v16h;

#if __has_builtin(__builtin_amdgcn_wmma_f32_16x16x4_f32)
#define USE_F32_WMMA 1
#else
#define USE_F32_WMMA 0
#endif

// ---------------- degree / normalization ----------------

__global__ void k_deg_self(float* deg, int n) {
    int i = blockIdx.x * blockDim.x + threadIdx.x;
    if (i < n) deg[i] = 1.0f;  // self loop contributes 1 to every node
}

__global__ void k_deg_edges(const long long* __restrict__ dst, float* deg, int e) {
    int i = blockIdx.x * blockDim.x + threadIdx.x;
    if (i < e) unsafeAtomicAdd(&deg[(int)dst[i]], 1.0f);
}

__global__ void k_deg_rsqrt(float* deg, int n) {
    int i = blockIdx.x * blockDim.x + threadIdx.x;
    if (i < n) deg[i] = rsqrtf(deg[i]);  // deg >= 1 always (self loops)
}

// ---------------- WMMA GEMM: C[:, cColBase : cColBase+16*NT] = A @ W ----------------
// A: [M, K] fp32 row-major.  W: [K, 16*NT] fp32 row-major.  C row stride ldC.
// One wave computes a 16-row x (16*NT)-col tile. M must be a multiple of 16
// (50000 = 3125*16). RELU applies relu to A elements (fused relu(h)).

template <int NT, int RELU>
__global__ void k_gemm_wmma(const float* __restrict__ A, const float* __restrict__ W,
                            float* __restrict__ C, int M, int K, int ldW, int ldC,
                            int cColBase) {
    const int wave    = threadIdx.x >> 5;
    const int lane    = threadIdx.x & 31;
    const int rowTile = blockIdx.x * (blockDim.x >> 5) + wave;
    if (rowTile * 16 >= M) return;  // wave-uniform: EXEC stays all-ones for WMMA

    const int n  = lane & 15;   // column within 16-wide tile / row within A tile
    const int hi = lane >> 4;   // half-wave group

    v8f acc[NT] = {};
    const float* arow = A + (size_t)(rowTile * 16 + n) * K;

#if USE_F32_WMMA
    // Full fp32 precision path: V_WMMA_F32_16X16X4_F32
    const int koff = 2 * hi;  // lanes 0-15: K=0,1 ; lanes 16-31: K=2,3
    for (int k0 = 0; k0 < K; k0 += 4) {
        float a0 = arow[k0 + koff], a1 = arow[k0 + koff + 1];
        if (RELU) { a0 = fmaxf(a0, 0.0f); a1 = fmaxf(a1, 0.0f); }
        v2f a;  a.x = a0;  a.y = a1;
#pragma unroll
        for (int t = 0; t < NT; ++t) {
            v2f b;
            b.x = W[(size_t)(k0 + koff)     * ldW + t * 16 + n];
            b.y = W[(size_t)(k0 + koff + 1) * ldW + t * 16 + n];
            acc[t] = __builtin_amdgcn_wmma_f32_16x16x4_f32(
                false, a, false, b, (short)0, acc[t], false, false);
        }
    }
#else
    // Fallback: f16 inputs, f32 accumulate (codegen-confirmed builtin)
    for (int k0 = 0; k0 < K; k0 += 32) {
        v16h a;
#pragma unroll
        for (int j = 0; j < 8; ++j) {
            int kk = k0 + ((j < 4) ? 0 : 16) + hi * 8 + 2 * (j & 3);
            float a0 = arow[kk], a1 = arow[kk + 1];
            if (RELU) { a0 = fmaxf(a0, 0.0f); a1 = fmaxf(a1, 0.0f); }
            a[2 * j] = (_Float16)a0;  a[2 * j + 1] = (_Float16)a1;
        }
#pragma unroll
        for (int t = 0; t < NT; ++t) {
            v16h b;
#pragma unroll
            for (int j = 0; j < 8; ++j) {
                int kk = k0 + ((j < 4) ? 0 : 16) + hi * 8 + 2 * (j & 3);
                b[2 * j]     = (_Float16)W[(size_t)kk       * ldW + t * 16 + n];
                b[2 * j + 1] = (_Float16)W[(size_t)(kk + 1) * ldW + t * 16 + n];
            }
            acc[t] = __builtin_amdgcn_wmma_f32_16x16x32_f16(
                false, a, false, b, (short)0, acc[t], false, false);
        }
    }
#endif

    // C/D layout: lane (n = lane&15) is column; VGPR v is row v + 8*hi
#pragma unroll
    for (int t = 0; t < NT; ++t) {
#pragma unroll
        for (int v = 0; v < 8; ++v) {
            int row = rowTile * 16 + hi * 8 + v;
            C[(size_t)row * ldC + cColBase + t * 16 + n] = acc[t][v];
        }
    }
}

// ---------------- aggregation ----------------

// out[i,f] = hp[i,f] * dinv[i]^2 (self-loop message) + bias[f]   (F == 128)
__global__ void k_agg_init(const float* __restrict__ hp, const float* __restrict__ dinv,
                           const float* __restrict__ bias, float* __restrict__ out,
                           int n) {
    int idx = blockIdx.x * blockDim.x + threadIdx.x;
    if (idx >= n * 128) return;
    int i = idx >> 7, f = idx & 127;
    float d = dinv[i];
    out[idx] = hp[idx] * d * d + bias[f];
}

// One wave32 per edge: out[dst] += hp[src] * dinv[src]*dinv[dst]   (F == 128)
__global__ void k_agg_edges(const float* __restrict__ hp,
                            const long long* __restrict__ src,
                            const long long* __restrict__ dst,
                            const float* __restrict__ dinv,
                            float* __restrict__ out, int e) {
    int wv   = (blockIdx.x * blockDim.x + threadIdx.x) >> 5;
    int lane = threadIdx.x & 31;
    if (wv >= e) return;  // wave-uniform
    int s = (int)src[wv];
    int d = (int)dst[wv];
    float nrm = dinv[s] * dinv[d];
    const float4* hrow = reinterpret_cast<const float4*>(hp + (size_t)s * 128);
    float* orow = out + (size_t)d * 128;
    float4 v = hrow[lane];  // one global_load_b128 covers 4 features per lane
    unsafeAtomicAdd(&orow[lane * 4 + 0], v.x * nrm);
    unsafeAtomicAdd(&orow[lane * 4 + 1], v.y * nrm);
    unsafeAtomicAdd(&orow[lane * 4 + 2], v.z * nrm);
    unsafeAtomicAdd(&orow[lane * 4 + 3], v.w * nrm);
}

__global__ void k_pack_bias(const float* __restrict__ bmu, const float* __restrict__ bls,
                            float* __restrict__ bcat) {
    int t = threadIdx.x;
    if (t < 64) bcat[t] = bmu[t];
    else if (t < 128) bcat[t] = bls[t - 64];
}

// d_out = [ mu (n x 64) flat | logstd (n x 64) flat ]
__global__ void k_split_out(const float* __restrict__ agg2, float* __restrict__ out, int n) {
    int idx = blockIdx.x * blockDim.x + threadIdx.x;
    if (idx >= n * 128) return;
    int i = idx >> 7, f = idx & 127;
    float v = agg2[idx];
    if (f < 64) out[(size_t)i * 64 + f] = v;
    else        out[(size_t)n * 64 + (size_t)i * 64 + (f - 64)] = v;
}

// ---------------- launch ----------------

extern "C" void kernel_launch(void* const* d_in, const int* in_sizes, int n_in,
                              void* d_out, int out_size, void* d_ws, size_t ws_size,
                              hipStream_t stream) {
    const int IN_C = 256;
    const int Nn   = in_sizes[0] / IN_C;  // 50000
    const int E    = in_sizes[1] / 2;     // 800000
    const int HID  = in_sizes[3];         // 128
    const int OUTC = in_sizes[5];         // 64
    (void)n_in; (void)out_size; (void)ws_size; (void)HID; (void)OUTC;

    const float*     x   = (const float*)d_in[0];
    const long long* ei  = (const long long*)d_in[1];  // int64 [2, E]
    const float*     W1  = (const float*)d_in[2];
    const float*     b1  = (const float*)d_in[3];
    const float*     Wmu = (const float*)d_in[4];
    const float*     bmu = (const float*)d_in[5];
    const float*     Wls = (const float*)d_in[6];
    const float*     bls = (const float*)d_in[7];
    const long long* src = ei;
    const long long* dst = ei + E;

    float* ws   = (float*)d_ws;
    float* dinv = ws;                         // N floats
    float* bcat = ws + Nn;                    // 128 floats
    float* buf0 = bcat + 128;                 // N*128 : h1_pre, then [h@Wmu | h@Wls]
    float* buf1 = buf0 + (size_t)Nn * 128;    // N*128 : agg1, then agg2

    float* out = (float*)d_out;

    const int T = 256;
    const int mTiles     = (Nn + 15) / 16;          // 3125
    const int gemmBlocks = (mTiles + 7) / 8;        // 8 waves per block
    const int edgeBlocks = (E + 7) / 8;             // 8 edges (waves) per block

    // degree -> dinv (in place)
    k_deg_self <<<(Nn + T - 1) / T, T, 0, stream>>>(dinv, Nn);
    k_deg_edges<<<(E  + T - 1) / T, T, 0, stream>>>(dst, dinv, E);
    k_deg_rsqrt<<<(Nn + T - 1) / T, T, 0, stream>>>(dinv, Nn);

    // Layer 1: h1_pre = x @ W1 (no bias yet)
    k_gemm_wmma<8, 0><<<gemmBlocks, T, 0, stream>>>(x, W1, buf0, Nn, IN_C, 128, 128, 0);
    // agg1 = scatter(h1_pre * norm) + self-loop + b1
    k_agg_init <<<(Nn * 128 + T - 1) / T, T, 0, stream>>>(buf0, dinv, b1, buf1, Nn);
    k_agg_edges<<<edgeBlocks, T, 0, stream>>>(buf0, src, dst, dinv, buf1, E);

    // Layer 2: [relu(agg1) @ Wmu | relu(agg1) @ Wls] -> buf0 (ReLU fused in A load)
    k_gemm_wmma<4, 1><<<gemmBlocks, T, 0, stream>>>(buf1, Wmu, buf0, Nn, 128, 64, 128, 0);
    k_gemm_wmma<4, 1><<<gemmBlocks, T, 0, stream>>>(buf1, Wls, buf0, Nn, 128, 64, 128, 64);

    // agg2 over combined 128-wide features (one edge pass for mu AND logstd)
    k_pack_bias<<<1, 128, 0, stream>>>(bmu, bls, bcat);
    k_agg_init <<<(Nn * 128 + T - 1) / T, T, 0, stream>>>(buf0, dinv, bcat, buf1, Nn);
    k_agg_edges<<<edgeBlocks, T, 0, stream>>>(buf0, src, dst, dinv, buf1, E);

    // split -> d_out = [mu | logstd]
    k_split_out<<<(Nn * 128 + T - 1) / T, T, 0, stream>>>(buf1, out, Nn);
}
